// SAViModel_37598143710092
// MI455X (gfx1250) — matrix operand order repaired and checked by reading, not compiled, not run on hardware
//
#include <hip/hip_runtime.h>

typedef __attribute__((ext_vector_type(16))) _Float16 v16h;
typedef __attribute__((ext_vector_type(8)))  _Float16 v8h;
typedef __attribute__((ext_vector_type(8)))  float    v8f;

#define WGT 256   // 8 waves of 32 (wave32) for the WMMA kernels
#define BM 64
#define BN 64
#define BK 64

// ---------------------------------------------------------------------------
// Fragment loads from LDS (all contiguous -> ds_load_b128 pairs).
// A layout (16x32 f16 per step): lane m=lane%16, K = 16*(j/8) + 8*(lane/16) + (j%8)
//   -> two contiguous 8-half runs at half-offsets {abase, abase+16}, abase=8*(lane/16)
// B tile stored transposed in LDS (Bst[n][k]): lane n=lane%16,
//   K = 16*(j/8) + 8*(lane/16) + (j%8) -> two 8-half runs at {bbase, bbase+16}, bbase=8*(lane/16)
// C/D layout: row = r + 8*(lane/16), col = lane%16
// ---------------------------------------------------------------------------
__device__ __forceinline__ v16h load_afrag(const _Float16* __restrict__ row, int abase)
{
  v8h a0 = *(const v8h*)(row + abase);
  v8h a1 = *(const v8h*)(row + 16 + abase);
  return __builtin_shufflevector(a0, a1, 0,1,2,3,4,5,6,7,8,9,10,11,12,13,14,15);
}
__device__ __forceinline__ v16h load_bfrag(const _Float16* __restrict__ col)
{
  v8h b0 = *(const v8h*)(col);
  v8h b1 = *(const v8h*)(col + 16);
  return __builtin_shufflevector(b0, b1, 0,1,2,3,4,5,6,7,8,9,10,11,12,13,14,15);
}

// ---------------------------------------------------------------------------
// Generic GEMM: C[b] = alpha * op(A[b]) x op(B[b]) (+ bias) (+ relu)
// tA: 0 -> A[m*K+k], 1 -> A[k*M+m];  tB: 0 -> B[k*N+n], 1 -> B[n*K+k]
// Block tile 64x64x64, 8 waves; wave (wm,wn) owns rows wm*32+{0,16}, cols wn*16.
// 4 v_wmma per LDS round per wave.
// ---------------------------------------------------------------------------
__global__ __launch_bounds__(WGT)
void gemm_wmma_k(const float* __restrict__ A, const float* __restrict__ Bm,
                 const float* __restrict__ bias, float* __restrict__ C,
                 int M, int N, int K, int tA, int tB,
                 long sA, long sB, long sC, float alpha, int relu)
{
  __shared__ _Float16 As[BM][BK];
  __shared__ _Float16 Bst[BN][BK];
  const int tid  = threadIdx.x;
  const int lane = tid & 31;
  const int wave = tid >> 5;
  const int wm = wave >> 2, wn = wave & 3;
  const int b  = blockIdx.z;
  const long m0 = (long)blockIdx.y * BM;
  const long n0 = (long)blockIdx.x * BN;
  const float* Ab = A  + (long)b * sA;
  const float* Bb = Bm + (long)b * sB;
  float*       Cb = C  + (long)b * sC;

  v8f acc0 = {0.f,0.f,0.f,0.f,0.f,0.f,0.f,0.f};
  v8f acc1 = {0.f,0.f,0.f,0.f,0.f,0.f,0.f,0.f};
  const int abase = (lane >> 4) << 3;
  const int bbase = (lane >> 4) << 3;
  const _Float16* arow0 = &As[wm * 32 + (lane & 15)][0];
  const _Float16* arow1 = &As[wm * 32 + 16 + (lane & 15)][0];
  const _Float16* bcol  = &Bst[wn * 16 + (lane & 15)][0];

  for (int k0 = 0; k0 < K; k0 += BK) {
    if (k0 + BK < K) __builtin_prefetch(Ab + (long)(k0 + BK), 0, 1);
    for (int e = 0; e < 16; ++e) {
      int idx = tid * 16 + e;          // 0..4095
      int mi = idx >> 6, kk = idx & 63;
      long m = m0 + mi; int k = k0 + kk;
      float va = 0.f;
      if (m < M && k < K)
        va = tA ? Ab[(long)k * M + m] : Ab[m * (long)K + k];
      As[mi][kk] = (_Float16)va;
      long n = n0 + kk; int k2 = k0 + mi;
      float vb = 0.f;
      if (n < N && k2 < K)
        vb = tB ? Bb[n * (long)K + k2] : Bb[(long)k2 * N + n];
      Bst[kk][mi] = (_Float16)vb;      // transposed store: [n][k]
    }
    __syncthreads();
#pragma unroll
    for (int ks = 0; ks < 2; ++ks) {
      v16h bf = load_bfrag(bcol + ks * 32 + bbase);
      v16h a0 = load_afrag(arow0 + ks * 32, abase);
      v16h a1 = load_afrag(arow1 + ks * 32, abase);
      acc0 = __builtin_amdgcn_wmma_f32_16x16x32_f16(false, a0, false, bf, (short)0, acc0, false, false);
      acc1 = __builtin_amdgcn_wmma_f32_16x16x32_f16(false, a1, false, bf, (short)0, acc1, false, false);
    }
    __syncthreads();
  }
  const long n = n0 + wn * 16 + (lane & 15);
  const int crow = wm * 32 + ((lane >> 4) << 3);
  float bi = 0.f;
  if (bias && n < N) bi = bias[n];
#pragma unroll
  for (int r = 0; r < 8; ++r) {
    long m = m0 + crow + r;
    if (m < M && n < N) {
      float v = alpha * acc0[r] + bi;
      if (relu) v = fmaxf(v, 0.f);
      Cb[m * (long)N + n] = v;
    }
    long m2 = m + 16;
    if (m2 < M && n < N) {
      float v = alpha * acc1[r] + bi;
      if (relu) v = fmaxf(v, 0.f);
      Cb[m2 * (long)N + n] = v;
    }
  }
}

// ---------------------------------------------------------------------------
// Implicit-GEMM 5x5 conv / transposed-conv (fractional stride gather).
// flipT=0: plain conv stride 1, W[oc][ic][ky][kx]
// flipT=1: conv_t: W'[oc][ic][ky][kx] = W[ic][oc][4-ky][4-kx], zero-dilated x
// ---------------------------------------------------------------------------
__global__ __launch_bounds__(WGT)
void conv_wmma_k(const float* __restrict__ X, const float* __restrict__ Wt,
                 const float* __restrict__ bias, float* __restrict__ Y,
                 int Cin, int Hin, int Win, int Cout, int Hout, int Wout,
                 int lo, int stride, int flipT, int relu, long imgStride)
{
  __shared__ _Float16 As[BM][BK];
  __shared__ _Float16 Bst[BN][BK];
  const int tid  = threadIdx.x;
  const int lane = tid & 31;
  const int wave = tid >> 5;
  const int wm = wave >> 2, wn = wave & 3;
  const int nImg = blockIdx.z;
  const float* Xb = X + (long)nImg * imgStride;
  const int Msz  = Hout * Wout;
  const int Ktot = Cin * 25;
  const long m0 = (long)blockIdx.y * BM;
  const long n0 = (long)blockIdx.x * BN;

  v8f acc0 = {0.f,0.f,0.f,0.f,0.f,0.f,0.f,0.f};
  v8f acc1 = {0.f,0.f,0.f,0.f,0.f,0.f,0.f,0.f};
  const int abase = (lane >> 4) << 3;
  const int bbase = (lane >> 4) << 3;
  const _Float16* arow0 = &As[wm * 32 + (lane & 15)][0];
  const _Float16* arow1 = &As[wm * 32 + 16 + (lane & 15)][0];
  const _Float16* bcol  = &Bst[wn * 16 + (lane & 15)][0];

  for (int k0 = 0; k0 < Ktot; k0 += BK) {
    if (k0 + BK < Ktot) __builtin_prefetch(Xb + (long)(k0 + BK), 0, 1);
    for (int e = 0; e < 16; ++e) {
      int idx = tid * 16 + e;
      int mi = idx >> 6, kk = idx & 63;
      long m = m0 + mi; int k = k0 + kk;
      float va = 0.f;
      if (m < Msz && k < Ktot) {
        int oy = (int)(m / Wout), ox = (int)(m % Wout);
        int ic = k / 25, kr = k % 25, ky = kr / 5, kx = kr % 5;
        int iyn = oy - lo + ky, ixn = ox - lo + kx;
        if (stride == 1) {
          if (iyn >= 0 && iyn < Hin && ixn >= 0 && ixn < Win)
            va = Xb[((long)ic * Hin + iyn) * Win + ixn];
        } else {
          if (iyn >= 0 && ixn >= 0 && (iyn % stride) == 0 && (ixn % stride) == 0) {
            int iy = iyn / stride, ix = ixn / stride;
            if (iy < Hin && ix < Win)
              va = Xb[((long)ic * Hin + iy) * Win + ix];
          }
        }
      }
      As[mi][kk] = (_Float16)va;
      int oc = (int)(n0 + kk); int k2 = k0 + mi;
      float vb = 0.f;
      if (oc < Cout && k2 < Ktot) {
        int ic = k2 / 25, kr = k2 % 25, ky = kr / 5, kx = kr % 5;
        vb = flipT ? Wt[(((long)ic * Cout + oc) * 25) + (4 - ky) * 5 + (4 - kx)]
                   : Wt[(((long)oc * Cin + ic) * 25) + ky * 5 + kx];
      }
      Bst[kk][mi] = (_Float16)vb;
    }
    __syncthreads();
#pragma unroll
    for (int ks = 0; ks < 2; ++ks) {
      v16h bf = load_bfrag(bcol + ks * 32 + bbase);
      v16h a0 = load_afrag(arow0 + ks * 32, abase);
      v16h a1 = load_afrag(arow1 + ks * 32, abase);
      acc0 = __builtin_amdgcn_wmma_f32_16x16x32_f16(false, a0, false, bf, (short)0, acc0, false, false);
      acc1 = __builtin_amdgcn_wmma_f32_16x16x32_f16(false, a1, false, bf, (short)0, acc1, false, false);
    }
    __syncthreads();
  }
  const long oc = n0 + wn * 16 + (lane & 15);
  const int crow = wm * 32 + ((lane >> 4) << 3);
  float bi = (oc < Cout) ? bias[oc] : 0.f;
#pragma unroll
  for (int r = 0; r < 8; ++r) {
    long m = m0 + crow + r;
    if (m < Msz && oc < Cout) {
      float v = acc0[r] + bi;
      if (relu) v = fmaxf(v, 0.f);
      Y[((long)nImg * Cout + oc) * Msz + m] = v;
    }
    long m2 = m + 16;
    if (m2 < Msz && oc < Cout) {
      float v = acc1[r] + bi;
      if (relu) v = fmaxf(v, 0.f);
      Y[((long)nImg * Cout + oc) * Msz + m2] = v;
    }
  }
}

// ---------------------------------------------------------------------------
// LayerNorm: one wave per row (wave32 shuffle reduction).
// ---------------------------------------------------------------------------
__global__ __launch_bounds__(32)
void ln_k(const float* __restrict__ x, const float* __restrict__ g,
          const float* __restrict__ bb, float* __restrict__ y, int L)
{
  const int row = blockIdx.x, lane = threadIdx.x;
  const float* xr = x + (long)row * L;
  float s = 0.f, sq = 0.f;
  for (int i = lane; i < L; i += 32) { float v = xr[i]; s += v; sq += v * v; }
  for (int o = 16; o > 0; o >>= 1) { s += __shfl_xor(s, o, 32); sq += __shfl_xor(sq, o, 32); }
  float m   = s / (float)L;
  float var = sq / (float)L - m * m;
  float inv = rsqrtf(var + 1e-5f);
  float* yr = y + (long)row * L;
  for (int i = lane; i < L; i += 32) yr[i] = (xr[i] - m) * inv * g[i] + bb[i];
}

// MHA core: per (b,h,q) block of 32 lanes (d=32); S=7 keys.
__global__ __launch_bounds__(32)
void mha_attn_k(const float* __restrict__ qkv, float* __restrict__ o)
{
  const int blk = blockIdx.x;
  const int h  = blk & 3;
  const int sq = (blk >> 2) % 7;
  const int b  = blk / 28;
  const int lane = threadIdx.x;
  const float qv = qkv[(long)(b * 7 + sq) * 384 + h * 32 + lane] * 0.17677669529663687f;
  float lg[7];
  for (int k = 0; k < 7; ++k) {
    float kv = qkv[(long)(b * 7 + k) * 384 + 128 + h * 32 + lane];
    float p  = qv * kv;
    for (int off = 16; off > 0; off >>= 1) p += __shfl_xor(p, off, 32);
    lg[k] = p;
  }
  float mx = lg[0];
  for (int k = 1; k < 7; ++k) mx = fmaxf(mx, lg[k]);
  float den = 0.f;
  for (int k = 0; k < 7; ++k) { lg[k] = __expf(lg[k] - mx); den += lg[k]; }
  float accv = 0.f;
  for (int k = 0; k < 7; ++k)
    accv += (lg[k] / den) * qkv[(long)(b * 7 + k) * 384 + 256 + h * 32 + lane];
  o[(long)(b * 7 + sq) * 128 + h * 32 + lane] = accv;
}

// softmax over S=7 slots + EPS (attn: [B,4096,7])
__global__ __launch_bounds__(256)
void softmax_s_k(float* __restrict__ attn)
{
  long i = (long)blockIdx.x * blockDim.x + threadIdx.x;
  if (i >= (long)8 * 4096) return;
  float* a = attn + i * 7;
  float mx = a[0];
  for (int s = 1; s < 7; ++s) mx = fmaxf(mx, a[s]);
  float e[7], den = 0.f;
  for (int s = 0; s < 7; ++s) { e[s] = __expf(a[s] - mx); den += e[s]; }
  for (int s = 0; s < 7; ++s) a[s] = e[s] / den + 1e-8f;
}

__global__ __launch_bounds__(256)
void attn_colsum_k(const float* __restrict__ attn, float* __restrict__ sums)
{
  __shared__ float red[256];
  const int bs = blockIdx.x, b = bs / 7, s = bs % 7;
  float acc = 0.f;
  for (int n = threadIdx.x; n < 4096; n += 256)
    acc += attn[((long)b * 4096 + n) * 7 + s];
  red[threadIdx.x] = acc; __syncthreads();
  for (int o = 128; o > 0; o >>= 1) {
    if ((int)threadIdx.x < o) red[threadIdx.x] += red[threadIdx.x + o];
    __syncthreads();
  }
  if (threadIdx.x == 0) sums[bs] = red[0];
}

__global__ __launch_bounds__(256)
void attn_norm_k(float* __restrict__ attn, const float* __restrict__ sums)
{
  long i = (long)blockIdx.x * blockDim.x + threadIdx.x;
  if (i >= (long)8 * 4096 * 7) return;
  int s = (int)(i % 7);
  int b = (int)((i / 7) >> 12);
  attn[i] /= sums[b * 7 + s];
}

__global__ __launch_bounds__(256)
void gru_combine_k(const float* __restrict__ gi, const float* __restrict__ gh,
                   const float* __restrict__ hp, float* __restrict__ out)
{
  long i = (long)blockIdx.x * blockDim.x + threadIdx.x;
  if (i >= 56L * 128) return;
  int row = (int)(i >> 7), c = (int)(i & 127);
  const float* gr = gi + (long)row * 384;
  const float* hr = gh + (long)row * 384;
  float r  = 1.f / (1.f + __expf(-(gr[c]       + hr[c])));
  float z  = 1.f / (1.f + __expf(-(gr[c + 128] + hr[c + 128])));
  float nn = tanhf(gr[c + 256] + r * hr[c + 256]);
  out[i] = (1.f - z) * nn + z * hp[i];
}

__global__ __launch_bounds__(256)
void add_kk(const float* __restrict__ a, const float* __restrict__ b,
            float* __restrict__ y, long n)
{
  long i = (long)blockIdx.x * blockDim.x + threadIdx.x;
  if (i < n) y[i] = a[i] + b[i];
}

__global__ __launch_bounds__(256)
void init_slots_k(const float* __restrict__ mu, const float* __restrict__ ls,
                  const float* __restrict__ noise, float* __restrict__ slots)
{
  long i = (long)blockIdx.x * blockDim.x + threadIdx.x;
  if (i >= 56L * 128) return;
  int d = (int)(i & 127);
  slots[i] = mu[d] + __expf(ls[d]) * noise[i];
}

// conv feats [B,64,64,64] + pos-embed -> tokens [B,4096,64]
__global__ __launch_bounds__(256)
void to_tokens_k(const float* __restrict__ x, const float* __restrict__ pw,
                 const float* __restrict__ pb, float* __restrict__ tok)
{
  long i = (long)blockIdx.x * blockDim.x + threadIdx.x;
  if (i >= 8L * 4096 * 64) return;
  int c = (int)(i & 63);
  long bp = i >> 6;
  int p = (int)(bp & 4095);
  int b = (int)(bp >> 12);
  float fy = (float)(p >> 6) * (1.f / 63.f);
  float fx = (float)(p & 63) * (1.f / 63.f);
  float pos = fy * pw[c] + fx * pw[64 + c] + (1.f - fy) * pw[128 + c]
            + (1.f - fx) * pw[192 + c] + pb[c];
  tok[i] = x[((long)b * 64 + c) * 4096 + p] + pos;
}

// broadcast slots [56,128] + dec pos-embed -> [56,128,8,8]
__global__ __launch_bounds__(256)
void bcast_dec_k(const float* __restrict__ slots, const float* __restrict__ pw,
                 const float* __restrict__ pb, float* __restrict__ d0)
{
  long i = (long)blockIdx.x * blockDim.x + threadIdx.x;
  if (i >= 56L * 128 * 64) return;
  int p = (int)(i & 63);
  long nc = i >> 6;
  int c = (int)(nc & 127);
  int n = (int)(nc >> 7);
  float fy = (float)(p >> 3) * (1.f / 7.f);
  float fx = (float)(p & 7) * (1.f / 7.f);
  float pos = fy * pw[c] + fx * pw[128 + c] + (1.f - fy) * pw[256 + c]
            + (1.f - fx) * pw[384 + c] + pb[c];
  d0[i] = slots[(long)n * 128 + c] + pos;
}

__global__ __launch_bounds__(256)
void store_slots_k(const float* __restrict__ s, float* __restrict__ out, int t)
{
  long i = (long)blockIdx.x * blockDim.x + threadIdx.x;
  if (i >= 56L * 128) return;
  int n = (int)(i >> 7), d = (int)(i & 127);
  out[((long)n * 6 + t) * 128 + d] = s[i];
}

// decoder frame [56,4,64,64] -> out_all [B,S,T,4,H,W]
__global__ __launch_bounds__(256)
void scatter_out_k(const float* __restrict__ d4, float* __restrict__ oa, int t)
{
  long i = (long)blockIdx.x * blockDim.x + threadIdx.x;
  if (i >= 56L * 4 * 4096) return;
  int p = (int)(i & 4095);
  long nc = i >> 12;
  int c = (int)(nc & 3);
  int n = (int)(nc >> 2);
  oa[(((long)n * 6 + t) * 4 + c) * 4096 + p] = d4[i];
}

// masks softmax over S, recons copy, weighted recon sum
__global__ __launch_bounds__(256)
void finalize_k(const float* __restrict__ oa, float* __restrict__ recon,
                float* __restrict__ recons, float* __restrict__ masks)
{
  __builtin_amdgcn_s_wait_tensorcnt(0);  // gfx1250 split-counter wait (no-op here)
  long i = (long)blockIdx.x * blockDim.x + threadIdx.x;
  if (i >= 8L * 6 * 4096) return;
  int p = (int)(i % 4096);
  int t = (int)((i / 4096) % 6);
  int b = (int)(i / (6 * 4096));
  float lg[7], mx = -1e30f;
  for (int s = 0; s < 7; ++s) {
    lg[s] = oa[(((long)(b * 7 + s) * 6 + t) * 4 + 3) * 4096 + p];
    mx = fmaxf(mx, lg[s]);
  }
  float den = 0.f;
  for (int s = 0; s < 7; ++s) { lg[s] = __expf(lg[s] - mx); den += lg[s]; }
  float rc0 = 0.f, rc1 = 0.f, rc2 = 0.f;
  for (int s = 0; s < 7; ++s) {
    float mk = lg[s] / den;
    long base = (long)(b * 7 + s) * 6 + t;
    masks[base * 4096 + p] = mk;
    float v0 = oa[(base * 4 + 0) * 4096 + p];
    float v1 = oa[(base * 4 + 1) * 4096 + p];
    float v2 = oa[(base * 4 + 2) * 4096 + p];
    recons[(base * 3 + 0) * 4096 + p] = v0;
    recons[(base * 3 + 1) * 4096 + p] = v1;
    recons[(base * 3 + 2) * 4096 + p] = v2;
    rc0 += v0 * mk; rc1 += v1 * mk; rc2 += v2 * mk;
  }
  long rb = ((long)b * 6 + t) * 3;
  recon[(rb + 0) * 4096 + p] = rc0;
  recon[(rb + 1) * 4096 + p] = rc1;
  recon[(rb + 2) * 4096 + p] = rc2;
}

// ---------------------------------------------------------------------------
extern "C" void kernel_launch(void* const* d_in, const int* in_sizes, int n_in,
                              void* d_out, int out_size, void* d_ws, size_t ws_size,
                              hipStream_t stream)
{
  (void)in_sizes; (void)out_size; (void)ws_size;
  const float* F[64];
  for (int i = 0; i < n_in && i < 64; ++i) F[i] = (const float*)d_in[i];
  // 0:batch 1:noise 2:mu 3:log_sigma 4-7:enc_w 8-11:enc_b 12:enc_pos_w 13:enc_pos_b
  // 14:eo_w1 15:eo_b1 16:eo_w2 17:eo_b2 18:ni_g 19:ni_b 20:ns_g 21:ns_b 22:nm_g 23:nm_b
  // 24:mha_in_w 25:mha_in_b 26:mha_out_w 27:mha_out_b 28:ma_w1 29:ma_b1 30:ma_w2 31:ma_b2
  // 32:Wq 33:Wk 34:Wv 35:gru_wih 36:gru_whh 37:gru_bih 38:gru_bhh
  // 39:mlp_w1 40:mlp_b1 41:mlp_w2 42:mlp_b2 43:dec_pos_w 44:dec_pos_b 45-48:dec_w 49-52:dec_b

  float* W0 = (float*)d_ws;
  size_t off = 0;
  auto alloc = [&](size_t n) { float* p = W0 + off; off += n; return p; };
  float* encA = alloc(2097152);
  float* encB = alloc(2097152);
  float* tok  = alloc(2097152);
  float* toka = alloc(4194304);
  float* tokb = alloc(4194304);
  float* lnin = alloc(4194304);
  float* k1   = alloc(4194304);
  float* v1   = alloc(4194304);
  float* attn = alloc(229376);
  float* sums = alloc(64);
  float* sA   = alloc(7168);
  float* sB   = alloc(7168);
  float* tmpA = alloc(7168);
  float* tmpB = alloc(7168);
  float* lnq  = alloc(7168);
  float* qbuf = alloc(7168);
  float* qkvb = alloc(21504);
  float* attb = alloc(7168);
  float* att2 = alloc(7168);
  float* gi   = alloc(21504);
  float* gh   = alloc(21504);
  float* updb = alloc(7168);
  float* slots_state = alloc(7168);
  float* d0   = alloc(458752);
  float* d1   = alloc(917504);
  float* d2   = alloc(3670016);
  float* d3   = alloc(14680064);
  float* d4b  = alloc(917504);
  float* out_all = alloc(5505024);

  float* out_recon  = (float*)d_out;
  float* out_recons = out_recon + 589824;
  float* out_masks  = out_recons + 4128768;
  float* out_slots  = out_masks + 1376256;

  auto blks = [](long n) { return dim3((unsigned)((n + 255) / 256)); };

  auto gemm = [&](const float* A, const float* Bm, const float* bias, float* C,
                  int M, int N, int K, int tA, int tB,
                  long sAs, long sBs, long sCs, int batch, float alpha, int relu) {
    dim3 g((unsigned)((N + BN - 1) / BN), (unsigned)((M + BM - 1) / BM), (unsigned)batch);
    gemm_wmma_k<<<g, dim3(WGT), 0, stream>>>(A, Bm, bias, C, M, N, K, tA, tB,
                                             sAs, sBs, sCs, alpha, relu);
  };
  auto conv = [&](const float* X, const float* Wt, const float* bias, float* Y,
                  int Ci, int Hi, int Wi, int Co, int Ho, int Wo,
                  int lo, int s, int flip, int relu, long imgStride, int nImg) {
    dim3 g((unsigned)((Co + BN - 1) / BN), (unsigned)((Ho * Wo + BM - 1) / BM), (unsigned)nImg);
    conv_wmma_k<<<g, dim3(WGT), 0, stream>>>(X, Wt, bias, Y, Ci, Hi, Wi, Co, Ho, Wo,
                                             lo, s, flip, relu, imgStride);
  };

  init_slots_k<<<blks(7168), 256, 0, stream>>>(F[2], F[3], F[1], slots_state);

  for (int t = 0; t < 6; ++t) {
    // -------- encoder: 4x conv5x5 + ReLU --------
    conv(F[0] + (long)t * 3 * 4096, F[4], F[8],  encA, 3, 64, 64, 64, 64, 64,
         2, 1, 0, 1, (long)6 * 3 * 4096, 8);
    conv(encA, F[5], F[9],  encB, 64, 64, 64, 64, 64, 64, 2, 1, 0, 1, (long)64 * 4096, 8);
    conv(encB, F[6], F[10], encA, 64, 64, 64, 64, 64, 64, 2, 1, 0, 1, (long)64 * 4096, 8);
    conv(encA, F[7], F[11], encB, 64, 64, 64, 64, 64, 64, 2, 1, 0, 1, (long)64 * 4096, 8);
    to_tokens_k<<<blks(2097152), 256, 0, stream>>>(encB, F[12], F[13], tok);
    gemm(tok,  F[14], F[15], toka, 32768, 128, 64,  0, 0, 0, 0, 0, 1, 1.f, 1);
    gemm(toka, F[16], F[17], tokb, 32768, 128, 128, 0, 0, 0, 0, 0, 1, 1.f, 0);

    // -------- corrector --------
    ln_k<<<32768, 32, 0, stream>>>(tokb, F[18], F[19], lnin, 128);
    gemm(lnin, F[33], nullptr, k1, 32768, 128, 128, 0, 0, 0, 0, 0, 1, 1.f, 0);
    gemm(lnin, F[34], nullptr, v1, 32768, 128, 128, 0, 0, 0, 0, 0, 1, 1.f, 0);
    ln_k<<<56, 32, 0, stream>>>(slots_state, F[20], F[21], sA, 128);

    if (t != 1) {  // MHA self-attention branch
      gemm(sA, F[24], F[25], qkvb, 56, 384, 128, 0, 1, 0, 0, 0, 1, 1.f, 0);
      mha_attn_k<<<224, 32, 0, stream>>>(qkvb, attb);
      gemm(attb, F[26], F[27], att2, 56, 128, 128, 0, 1, 0, 0, 0, 1, 1.f, 0);
      add_kk<<<blks(7168), 256, 0, stream>>>(sA, att2, tmpA, 7168);
      ln_k<<<56, 32, 0, stream>>>(tmpA, F[20], F[21], sB, 128);
      gemm(sB,   F[28], F[29], tmpA, 56, 128, 128, 0, 0, 0, 0, 0, 1, 1.f, 1);
      gemm(tmpA, F[30], F[31], tmpB, 56, 128, 128, 0, 0, 0, 0, 0, 1, 1.f, 0);
      add_kk<<<blks(7168), 256, 0, stream>>>(tmpB, sB, sA, 7168);
    }

    for (int it = 0; it < 2; ++it) {  // NIT slot-attention iterations
      ln_k<<<56, 32, 0, stream>>>(sA, F[20], F[21], lnq, 128);
      gemm(lnq, F[32], nullptr, qbuf, 56, 128, 128, 0, 0, 0, 0, 0, 1, 1.f, 0);
      gemm(k1, qbuf, nullptr, attn, 4096, 7, 128, 0, 1,
           (long)4096 * 128, (long)7 * 128, (long)4096 * 7, 8, 0.08838834764831845f, 0);
      softmax_s_k<<<blks(32768), 256, 0, stream>>>(attn);
      attn_colsum_k<<<56, 256, 0, stream>>>(attn, sums);
      attn_norm_k<<<blks(229376), 256, 0, stream>>>(attn, sums);
      gemm(attn, v1, nullptr, updb, 7, 128, 4096, 1, 0,
           (long)4096 * 7, (long)4096 * 128, (long)7 * 128, 8, 1.f, 0);
      gemm(updb, F[35], F[37], gi, 56, 384, 128, 0, 1, 0, 0, 0, 1, 1.f, 0);
      gemm(sA,   F[36], F[38], gh, 56, 384, 128, 0, 1, 0, 0, 0, 1, 1.f, 0);
      gru_combine_k<<<blks(7168), 256, 0, stream>>>(gi, gh, sA, sB);
      ln_k<<<56, 32, 0, stream>>>(sB, F[22], F[23], lnq, 128);
      gemm(lnq,  F[39], F[40], tmpA, 56, 128, 128, 0, 0, 0, 0, 0, 1, 1.f, 1);
      gemm(tmpA, F[41], F[42], tmpB, 56, 128, 128, 0, 0, 0, 0, 0, 1, 1.f, 0);
      add_kk<<<blks(7168), 256, 0, stream>>>(sB, tmpB, sA, 7168);
    }
    hipMemcpyAsync(slots_state, sA, 7168 * sizeof(float),
                   hipMemcpyDeviceToDevice, stream);
    store_slots_k<<<blks(7168), 256, 0, stream>>>(sA, out_slots, t);

    // -------- decoder: 3x convT stride2 + final convT stride1 --------
    bcast_dec_k<<<blks(458752), 256, 0, stream>>>(sA, F[43], F[44], d0);
    conv(d0, F[45], F[49], d1, 128, 8,  8,  64, 16, 16, 2, 2, 1, 1, (long)128 * 64, 56);
    conv(d1, F[46], F[50], d2, 64, 16, 16, 64, 32, 32, 2, 2, 1, 1, (long)64 * 256, 56);
    conv(d2, F[47], F[51], d3, 64, 32, 32, 64, 64, 64, 2, 2, 1, 1, (long)64 * 1024, 56);
    conv(d3, F[48], F[52], d4b, 64, 64, 64, 4, 64, 64, 2, 1, 1, 0, (long)64 * 4096, 56);
    scatter_out_k<<<blks(917504), 256, 0, stream>>>(d4b, out_all, t);
  }

  finalize_k<<<blks(196608), 256, 0, stream>>>(out_all, out_recon, out_recons, out_masks);
}
